// MultiHeadedAttention_39728447488473
// MI455X (gfx1250) — matrix operand order repaired and checked by reading, not compiled, not run on hardware
//
#include <hip/hip_runtime.h>

typedef __attribute__((ext_vector_type(16))) __bf16 v16bf;
typedef __attribute__((ext_vector_type(8)))  float  v8f;

#define D_MODEL   1024
#define NUM_HEADS 16
#define HEAD_DIM  64
#define SEQ       2048
#define BATCH     2
#define M_TOTAL   (BATCH * SEQ)

#define BM 128
#define BN 128
#define BK 32
#define BQ 128
#define BKV 128

__device__ __forceinline__ unsigned short f32_to_bf16(float f) {
  union { float f; unsigned u; } v; v.f = f;
  unsigned x = v.u;
  unsigned r = x + 0x7FFFu + ((x >> 16) & 1u);  // round-to-nearest-even
  return (unsigned short)(r >> 16);
}

__device__ __forceinline__ v8f wmma_bf16(v16bf a, v16bf b, v8f c) {
  return __builtin_amdgcn_wmma_f32_16x16x32_bf16(
      false, a, false, b, (short)0, c, false, false);
}

// Async DMA: 16B/lane global -> LDS, tracked by ASYNCcnt (gfx1250).
// GVS mode: saddr(64b SGPR base) + vaddr(32b per-lane byte offset); vdst VGPR
// holds the per-lane LDS byte address (low 32 bits of generic shared address).
__device__ __forceinline__ void async_copy_b128(const void* gbase,
                                                unsigned voff_bytes,
                                                void* lds_ptr) {
  unsigned lds = (unsigned)(size_t)lds_ptr;
  unsigned long long sbase = (unsigned long long)(size_t)gbase;
  asm volatile("global_load_async_to_lds_b128 %0, %1, %2"
               :: "v"(lds), "v"(voff_bytes), "s"(sbase)
               : "memory");
}

__device__ __forceinline__ void async_wait0() {
#if __has_builtin(__builtin_amdgcn_s_wait_asynccnt)
  __builtin_amdgcn_s_wait_asynccnt(0);
#else
  asm volatile("s_wait_asynccnt 0x0" ::: "memory");
#endif
}

// A-fragment (16x32 bf16, row-major LDS tile, K contiguous, stride in elems).
// Lanes 0-15 M=lane, lanes 16-31 M=lane-16; K base = half*8 and 16+half*8.
__device__ __forceinline__ v16bf load_frag_a(const unsigned short* base, int stride, int lane) {
  const int hf = (lane >> 4) & 1;
  const int r  = lane & 15;
  const unsigned short* p = base + r * stride + hf * 8;
  union { uint4 q[2]; v16bf v; } u;
  u.q[0] = *(const uint4*)(p);       // K = hf*8 .. hf*8+7
  u.q[1] = *(const uint4*)(p + 16);  // K = 16+hf*8 .. 16+hf*8+7
  return u.v;
}

// B-fragment (32x16 bf16) from an N-major LDS tile (per column, K contiguous).
// lane&15 = N column; K base = half*8 and 16+half*8 (matches A-fragment).
__device__ __forceinline__ v16bf load_frag_b(const unsigned short* base, int stride, int lane) {
  const int hf = (lane >> 4) & 1;
  const int n  = lane & 15;
  const unsigned short* p = base + n * stride + hf * 8;
  union { uint4 q[2]; v16bf v; } u;
  u.q[0] = *(const uint4*)(p);
  u.q[1] = *(const uint4*)(p + 16);
  return u.v;
}

// ---------------------------------------------------------------------------
// QKV projection: OUT[b,h,s,e] (bf16) = X[m,:] @ W[h,:,e] + bias[h,e]
// X: [4096,1024] fp32 row-major; W: [H,K,E] fp32; m = b*SEQ+s, n = h*64+e.
// (fp32 -> bf16 conversion happens in flight, so this stays on the VGPR path)
// ---------------------------------------------------------------------------
__global__ __launch_bounds__(256) void proj_gemm_bf16(
    const float* __restrict__ X, const float* __restrict__ W,
    const float* __restrict__ bias, unsigned short* __restrict__ OUT) {
  __shared__ alignas(16) unsigned short As[BM][BK];
  __shared__ alignas(16) unsigned short Bs[BN][BK];

  const int tid  = threadIdx.x;
  const int lane = tid & 31;
  const int wave = tid >> 5;
  const int wm = wave >> 1, wn = wave & 1;     // 4x2 wave grid
  const int bm0 = blockIdx.x * BM;
  const int bn0 = blockIdx.y * BN;

  v8f acc[2][4];
  for (int i = 0; i < 2; ++i)
    for (int j = 0; j < 4; ++j)
      for (int e = 0; e < 8; ++e) acc[i][j][e] = 0.0f;

  for (int k0 = 0; k0 < D_MODEL; k0 += BK) {
    // A tile: 128x32 fp32 -> bf16 (float4 per thread x4)
    for (int i = 0; i < 4; ++i) {
      int f = tid + i * 256;                 // 1024 float4 slots
      int row = f >> 3, kq = (f & 7) * 4;
      float4 v = *(const float4*)(X + (size_t)(bm0 + row) * D_MODEL + k0 + kq);
      As[row][kq + 0] = f32_to_bf16(v.x);
      As[row][kq + 1] = f32_to_bf16(v.y);
      As[row][kq + 2] = f32_to_bf16(v.z);
      As[row][kq + 3] = f32_to_bf16(v.w);
    }
    // B tile: Bs[n][k] = W[h][k][e]; iterate e-fastest so global loads coalesce
    for (int i = 0; i < 16; ++i) {
      int f = tid + i * 256;                 // 4096 elems
      int nl = f & 127, kl = f >> 7;
      int n = bn0 + nl, k = k0 + kl;
      int h = n >> 6, e = n & 63;
      Bs[nl][kl] = f32_to_bf16(W[((size_t)h * D_MODEL + k) * HEAD_DIM + e]);
    }
    __syncthreads();

    v16bf a0 = load_frag_a(&As[wm * 32][0], BK, lane);
    v16bf a1 = load_frag_a(&As[wm * 32 + 16][0], BK, lane);
    for (int nt = 0; nt < 4; ++nt) {
      v16bf bb = load_frag_b(&Bs[wn * 64 + nt * 16][0], BK, lane);
      acc[0][nt] = wmma_bf16(a0, bb, acc[0][nt]);
      acc[1][nt] = wmma_bf16(a1, bb, acc[1][nt]);
    }
    __syncthreads();
  }

  const int hf = (lane >> 4) & 1;
  for (int mi = 0; mi < 2; ++mi)
    for (int nt = 0; nt < 4; ++nt)
      for (int j = 0; j < 8; ++j) {
        int row = bm0 + wm * 32 + mi * 16 + hf * 8 + j;
        int col = bn0 + wn * 64 + nt * 16 + (lane & 15);
        float v = acc[mi][nt][j] + bias[col];
        int b = row >> 11, s = row & (SEQ - 1);
        int h = col >> 6, e = col & 63;
        OUT[((size_t)(b * NUM_HEADS + h) * SEQ + s) * HEAD_DIM + e] = f32_to_bf16(v);
      }
}

// ---------------------------------------------------------------------------
// Flash attention: one block per (b, h, 128-row Q tile). bf16 in, bf16 ctx out
// in [B,S,H*E] layout. Q/K tiles staged via async global->LDS DMA.
// ---------------------------------------------------------------------------
__global__ __launch_bounds__(256) void flash_attn(
    const unsigned short* __restrict__ qh, const unsigned short* __restrict__ kh,
    const unsigned short* __restrict__ vh, unsigned short* __restrict__ ctx) {
  __shared__ alignas(16) unsigned short Qs[BQ][HEAD_DIM];
  __shared__ alignas(16) unsigned short Ks[BKV][HEAD_DIM];
  __shared__ alignas(16) unsigned short Vst[HEAD_DIM][BKV];  // transposed V
  __shared__ alignas(16) unsigned short Ps[8][16][BKV];      // per-wave P tile

  const int tid = threadIdx.x, lane = tid & 31, wave = tid >> 5;
  const int qt = blockIdx.x, h = blockIdx.y, b = blockIdx.z;
  const size_t head_base = (size_t)(b * NUM_HEADS + h) * SEQ * HEAD_DIM;
  const unsigned short* Q = qh + head_base;
  const unsigned short* K = kh + head_base;
  const unsigned short* V = vh + head_base;
  const int q0 = qt * BQ;

  // Q tile: async DMA, 16B per lane x4
  for (int i = 0; i < 4; ++i) {
    int f = tid + i * 256;
    int row = f >> 3, cq = (f & 7) * 8;
    async_copy_b128(Q, (unsigned)(((q0 + row) * HEAD_DIM + cq) * 2), &Qs[row][cq]);
  }
  async_wait0();
  __syncthreads();

  v16bf aq0 = load_frag_a(&Qs[wave * 16][0], HEAD_DIM, lane);
  v16bf aq1 = load_frag_a(&Qs[wave * 16][0] + 32, HEAD_DIM, lane);

  float m_run[8], l_run[8];
  v8f o_acc[4];
  for (int j = 0; j < 8; ++j) { m_run[j] = -1e30f; l_run[j] = 0.0f; }
  for (int et = 0; et < 4; ++et)
    for (int j = 0; j < 8; ++j) o_acc[et][j] = 0.0f;

  const float scale = 0.125f;  // 1/sqrt(64)
  const int hf = (lane >> 4) & 1;

  for (int kv0 = 0; kv0 < SEQ; kv0 += BKV) {
    __syncthreads();  // previous iteration done with Ks/Vst/Ps
    // K tile: async DMA
    for (int i = 0; i < 4; ++i) {
      int f = tid + i * 256;
      int row = f >> 3, cq = (f & 7) * 8;
      async_copy_b128(K, (unsigned)(((kv0 + row) * HEAD_DIM + cq) * 2), &Ks[row][cq]);
    }
    // V transposed into LDS (transform in flight -> VGPR path)
    for (int i = 0; i < 32; ++i) {
      int f = tid + i * 256;
      int t = f >> 6, e = f & 63;
      Vst[e][t] = V[(size_t)(kv0 + t) * HEAD_DIM + e];
    }
    async_wait0();
    __syncthreads();

    // S = Q * K^T  (16 q-rows per wave x 128 keys)
    v8f s_acc[8];
    for (int nt = 0; nt < 8; ++nt) {
      for (int j = 0; j < 8; ++j) s_acc[nt][j] = 0.0f;
      v16bf b0 = load_frag_b(&Ks[nt * 16][0], HEAD_DIM, lane);
      v16bf b1 = load_frag_b(&Ks[nt * 16][0] + 32, HEAD_DIM, lane);
      s_acc[nt] = wmma_bf16(aq0, b0, s_acc[nt]);
      s_acc[nt] = wmma_bf16(aq1, b1, s_acc[nt]);
    }

    // Online softmax: row = hf*8 + j, 16 lanes (one half-wave) share a row.
    for (int j = 0; j < 8; ++j) {
      float mx = -1e30f;
      for (int nt = 0; nt < 8; ++nt) mx = fmaxf(mx, s_acc[nt][j] * scale);
      for (int off = 1; off < 16; off <<= 1) mx = fmaxf(mx, __shfl_xor(mx, off, 32));
      float m_new = fmaxf(m_run[j], mx);
      float corr = __expf(m_run[j] - m_new);
      m_run[j] = m_new;

      float rsum = 0.0f;
      const int prow = hf * 8 + j;
      for (int nt = 0; nt < 8; ++nt) {
        float p = __expf(s_acc[nt][j] * scale - m_new);
        rsum += p;
        Ps[wave][prow][nt * 16 + (lane & 15)] = f32_to_bf16(p);
      }
      for (int off = 1; off < 16; off <<= 1) rsum += __shfl_xor(rsum, off, 32);
      l_run[j] = l_run[j] * corr + rsum;
      for (int et = 0; et < 4; ++et) o_acc[et][j] *= corr;
    }
    __syncthreads();  // P visible in LDS

    // O += P * V   (K dim = 128 keys, 4 WMMA k-steps)
    for (int kk = 0; kk < 4; ++kk) {
      v16bf ap = load_frag_a(&Ps[wave][0][0] + kk * 32, BKV, lane);
      for (int et = 0; et < 4; ++et) {
        v16bf bv = load_frag_b(&Vst[et * 16][0] + kk * 32, BKV, lane);
        o_acc[et] = wmma_bf16(ap, bv, o_acc[et]);
      }
    }
  }

  for (int j = 0; j < 8; ++j) {
    float inv = 1.0f / l_run[j];
    int row = q0 + wave * 16 + hf * 8 + j;
    for (int et = 0; et < 4; ++et) {
      int col = h * HEAD_DIM + et * 16 + (lane & 15);
      ctx[((size_t)b * SEQ + row) * D_MODEL + col] = f32_to_bf16(o_acc[et][j] * inv);
    }
  }
}

// ---------------------------------------------------------------------------
// Output projection: OUT[m,n] (fp32) = ctx[m,:] @ Wo[n,:] + bo[n]
// A tile (bf16 straight copy) staged via async global->LDS DMA.
// ---------------------------------------------------------------------------
__global__ __launch_bounds__(256) void out_gemm(
    const unsigned short* __restrict__ CTX, const float* __restrict__ Wo,
    const float* __restrict__ bo, float* __restrict__ OUT) {
  __shared__ alignas(16) unsigned short As[BM][BK];
  __shared__ alignas(16) unsigned short Bs[BN][BK];

  const int tid  = threadIdx.x;
  const int lane = tid & 31;
  const int wave = tid >> 5;
  const int wm = wave >> 1, wn = wave & 1;
  const int bm0 = blockIdx.x * BM;
  const int bn0 = blockIdx.y * BN;

  v8f acc[2][4];
  for (int i = 0; i < 2; ++i)
    for (int j = 0; j < 4; ++j)
      for (int e = 0; e < 8; ++e) acc[i][j][e] = 0.0f;

  for (int k0 = 0; k0 < D_MODEL; k0 += BK) {
    for (int i = 0; i < 2; ++i) {          // A tile: bf16 async DMA copy
      int f = tid + i * 256;               // 512 uint4 slots
      int row = f >> 2, kq = (f & 3) * 8;
      async_copy_b128(CTX, (unsigned)((((bm0 + row) * D_MODEL) + k0 + kq) * 2),
                      &As[row][kq]);
    }
    for (int i = 0; i < 16; ++i) {         // B(k,n) = Wo[n*K + k], k-coalesced
      int f = tid + i * 256;
      int nl = f >> 5, kl = f & 31;
      Bs[nl][kl] = f32_to_bf16(Wo[(size_t)(bn0 + nl) * D_MODEL + k0 + kl]);
    }
    async_wait0();
    __syncthreads();

    v16bf a0 = load_frag_a(&As[wm * 32][0], BK, lane);
    v16bf a1 = load_frag_a(&As[wm * 32 + 16][0], BK, lane);
    for (int nt = 0; nt < 4; ++nt) {
      v16bf bb = load_frag_b(&Bs[wn * 64 + nt * 16][0], BK, lane);
      acc[0][nt] = wmma_bf16(a0, bb, acc[0][nt]);
      acc[1][nt] = wmma_bf16(a1, bb, acc[1][nt]);
    }
    __syncthreads();
  }

  const int hf = (lane >> 4) & 1;
  for (int mi = 0; mi < 2; ++mi)
    for (int nt = 0; nt < 4; ++nt)
      for (int j = 0; j < 8; ++j) {
        int row = bm0 + wm * 32 + mi * 16 + hf * 8 + j;
        int col = bn0 + wn * 64 + nt * 16 + (lane & 15);
        OUT[(size_t)row * D_MODEL + col] = acc[mi][nt][j] + bo[col];
      }
}

extern "C" void kernel_launch(void* const* d_in, const int* in_sizes, int n_in,
                              void* d_out, int out_size, void* d_ws, size_t ws_size,
                              hipStream_t stream) {
  const float* q  = (const float*)d_in[0];
  const float* k  = (const float*)d_in[1];
  const float* v  = (const float*)d_in[2];
  const float* Wq = (const float*)d_in[3];
  const float* bq = (const float*)d_in[4];
  const float* Wk = (const float*)d_in[5];
  const float* bk = (const float*)d_in[6];
  const float* Wv = (const float*)d_in[7];
  const float* bv = (const float*)d_in[8];
  const float* Wo = (const float*)d_in[9];
  const float* bo = (const float*)d_in[10];

  const size_t nhe = (size_t)BATCH * NUM_HEADS * SEQ * HEAD_DIM;  // 4,194,304
  unsigned short* qh  = (unsigned short*)d_ws;
  unsigned short* kh  = qh + nhe;
  unsigned short* vh  = kh + nhe;
  unsigned short* ctx = vh + nhe;   // [B,S,1024] bf16; total ws: 32 MB

  dim3 gridG(M_TOTAL / BM, D_MODEL / BN, 1);  // (32, 8)
  proj_gemm_bf16<<<gridG, dim3(256), 0, stream>>>(q, Wq, bq, qh);
  proj_gemm_bf16<<<gridG, dim3(256), 0, stream>>>(k, Wk, bk, kh);
  proj_gemm_bf16<<<gridG, dim3(256), 0, stream>>>(v, Wv, bv, vh);
  flash_attn<<<dim3(SEQ / BQ, NUM_HEADS, BATCH), dim3(256), 0, stream>>>(qh, kh, vh, ctx);
  out_gemm<<<gridG, dim3(256), 0, stream>>>(ctx, Wo, bo, (float*)d_out);
}
